// STU_35098472743128
// MI455X (gfx1250) — matrix operand order repaired and checked by reading, not compiled, not run on hardware
//
#include <hip/hip_runtime.h>

// ---------------------------------------------------------------------------
// STU forward for MI455X (gfx1250): Toeplitz-GEMM conv + fused spectral/AR
// GEMM via v_wmma_f32_16x16x32_bf16; output-AR recurrence distributed over
// 32 workgroups with per-timestep release/acquire flag synchronization.
// Round 4: time-reversed filter table so conv A fragments are plain b128
// loads (no per-element half-swaps in the inner loop).
// ---------------------------------------------------------------------------

typedef __attribute__((ext_vector_type(16))) __bf16 v16bf;
typedef __attribute__((ext_vector_type(8)))  __bf16 v8bf;
typedef __attribute__((ext_vector_type(8)))  float  v8f;

#define B_   2
#define L_   1024
#define LP_  (L_ + 64)              // reversed filter + 64 trailing zeros
#define D_   512
#define K_   16
#define KU_  3
#define KY_  2
#define KD2_ (2 * K_ * D_)          // 16384
#define C_   (KD2_ + KU_ * D_)      // 17920 (contraction dim of fused GEMM)
#define CSPLIT_ 4                   // k_spectral contraction split
#define CCHUNK_ (C_ / CSPLIT_)      // 4480 (multiple of 32)
#define SPLITS_ 16                  // scan workgroups per batch
#define ROWS_   (D_ / SPLITS_)      // 32 output rows per scan workgroup

static __device__ __forceinline__ v8f wmma_bf16(v16bf a, v16bf b, v8f c) {
  return __builtin_amdgcn_wmma_f32_16x16x32_bf16(
      false, a, false, b, (short)0, c, false, false);
}

static __device__ __forceinline__ v16bf cat8(v8bf lo, v8bf hi) {
  return __builtin_shufflevector(lo, hi, 0, 1, 2, 3, 4, 5, 6, 7,
                                 8, 9, 10, 11, 12, 13, 14, 15);
}

// ------------------------------- prep kernels ------------------------------

// ubT[b][d][t] = bf16(u[b][t][d])   (transposed for WMMA B fragments)
__global__ void k_prep_ubT(const float* __restrict__ u, __bf16* __restrict__ ubT) {
  int idx = blockIdx.x * blockDim.x + threadIdx.x;
  if (idx >= B_ * L_ * D_) return;
  int t = idx % L_;
  int d = (idx / L_) % D_;
  int b = idx / (L_ * D_);
  ubT[idx] = (__bf16)u[(size_t)(b * L_ + t) * D_ + d];
}

// Time-reversed signed filter: frev[ck][i] = sgn_ck(s)*filters[s][ck%K] with
// s = 1023 - i (i <= 1023), and frev[ck][1024..1087] = 0 (causal padding).
__global__ void k_prep_frev(const float* __restrict__ filters, __bf16* __restrict__ frev) {
  int i = blockIdx.x * blockDim.x + threadIdx.x;
  if (i >= 2 * K_ * LP_) return;
  int p  = i % LP_;
  int ck = i / LP_;
  int k  = ck & (K_ - 1);
  int s  = (L_ - 1) - p;
  float v = 0.0f;
  if (s >= 0) {
    v = filters[s * K_ + k];
    if ((ck >= K_) && (s & 1)) v = -v;
  }
  frev[i] = (__bf16)v;
}

// McatT[o][k*D+d] = bf16(M_phi_plus[k][d][o]);  McatT[o][K*D+k*D+d] = minus
__global__ void k_prep_mphiT(const float* __restrict__ mp, const float* __restrict__ mm,
                             __bf16* __restrict__ McatT) {
  int idx = blockIdx.x * blockDim.x + threadIdx.x;
  if (idx >= K_ * D_ * D_) return;       // (k, d, o)
  int o = idx % D_;
  int d = (idx / D_) % D_;
  int k = idx / (D_ * D_);
  size_t row = (size_t)o * C_;
  McatT[row + k * D_ + d]            = (__bf16)mp[idx];
  McatT[row + K_ * D_ + k * D_ + d]  = (__bf16)mm[idx];
}

// McatT[i][2KD + j*D + f] = bf16(M_u[i][j][f])  (contiguous both sides)
__global__ void k_prep_muT(const float* __restrict__ Mu, __bf16* __restrict__ McatT) {
  int idx = blockIdx.x * blockDim.x + threadIdx.x;
  if (idx >= D_ * KU_ * D_) return;
  int rem = idx % (KU_ * D_);
  int i   = idx / (KU_ * D_);
  McatT[(size_t)i * C_ + KD2_ + rem] = (__bf16)Mu[idx];
}

// Ucat columns [2KD, C): shifted copies of u (shift j = 0..2, zero-filled)
__global__ void k_fill_shift(const float* __restrict__ u, __bf16* __restrict__ Ucat) {
  int idx = blockIdx.x * blockDim.x + threadIdx.x;
  if (idx >= B_ * L_ * KU_ * D_) return;
  int f = idx % D_;
  int j = (idx / D_) % KU_;
  int t = (idx / (D_ * KU_)) % L_;
  int b = idx / (D_ * KU_ * L_);
  int ts = t - j;
  float v = u[(size_t)(b * L_ + (ts < 0 ? 0 : ts)) * D_ + f];
  if (ts < 0) v = 0.0f;
  Ucat[(size_t)(b * L_ + t) * C_ + KD2_ + j * D_ + f] = (__bf16)v;
}

// --------------------- Toeplitz conv as WMMA GEMM --------------------------
// U[b,t,ck,d] = sum_{r<=t} f_ck[t-r] * u[b,r,d]; one wave = 16(t) x 64(d).
__global__ void k_conv(const __bf16* __restrict__ ubT, const __bf16* __restrict__ frev,
                       __bf16* __restrict__ Ucat) {
  int w    = blockIdx.x * 8 + (threadIdx.x >> 5);
  int lane = threadIdx.x & 31;
  int m    = lane & 15;
  int half = lane >> 4;

  int d64 = w & 7;          // D/64 tiles
  int t16 = (w >> 3) & 63;  // L/16 tiles
  int ck  = (w >> 9) & 31;  // 2K filters (sign-major concat)
  int b   = w >> 14;        // batch
  int t0 = t16 * 16, d0 = d64 * 64;

  // A[m, kk] = f_ck[(t0+m) - (r0+kk)] = frev[(1023-(t0+m)) + r0 + kk];
  // trailing zeros of frev implement the causal mask.  Index <= 1069 < LP.
  const __bf16* fr = frev + (size_t)ck * LP_ + ((L_ - 1) - (t0 + m));
  v8f acc[4] = {v8f{}, v8f{}, v8f{}, v8f{}};

  const int rmax = t0 + 15;
  for (int r0 = 0; r0 <= rmax; r0 += 32) {
    // kk ascending in e -> two contiguous 8-element loads (same pattern as B)
    v16bf a = cat8(*(const v8bf*)(fr + r0 + half * 8),
                   *(const v8bf*)(fr + r0 + 16 + half * 8));
#pragma unroll
    for (int j = 0; j < 4; ++j) {
      const __bf16* col = ubT + ((size_t)(b * D_ + d0 + j * 16 + m)) * L_
                              + r0 + half * 8;         // r <= 1023 always
      v16bf bm = cat8(*(const v8bf*)col, *(const v8bf*)(col + 16));
      acc[j] = wmma_bf16(a, bm, acc[j]);
    }
  }

#pragma unroll
  for (int j = 0; j < 4; ++j)
#pragma unroll
    for (int i = 0; i < 8; ++i) {
      int t = t0 + i + half * 8;                       // ISA f32 C/D layout
      int d = d0 + j * 16 + m;
      Ucat[(size_t)(b * L_ + t) * C_ + ck * D_ + d] = (__bf16)acc[j][i];
    }
}

// ------------- fused spectral projection + input-AR GEMM -------------------
// zpart[cs][b,t,o] = sum_{c in chunk cs} Ucat[b,t,c] * McatT[o,c]
// one wave = 16(t) x 64(o) x (C/4 chunk);  B*4*64*8 = 4096 waves.
__global__ void k_spectral(const __bf16* __restrict__ Ucat, const __bf16* __restrict__ McatT,
                           float* __restrict__ zpart) {
  int w    = blockIdx.x * 8 + (threadIdx.x >> 5);
  int lane = threadIdx.x & 31;
  int m    = lane & 15;
  int half = lane >> 4;

  int o64 = w & 7;
  int t16 = (w >> 3) & 63;
  int cs  = (w >> 9) & (CSPLIT_ - 1);
  int b   = w >> 11;
  int t0 = t16 * 16, o0 = o64 * 64;
  int cbeg = cs * CCHUNK_, cend = cbeg + CCHUNK_;

  v8f acc[4] = {v8f{}, v8f{}, v8f{}, v8f{}};
  const __bf16* arow = Ucat + (size_t)(b * L_ + t0 + m) * C_;

  for (int c0 = cbeg; c0 < cend; c0 += 32) {
    v16bf a = cat8(*(const v8bf*)(arow + c0 + half * 8),
                   *(const v8bf*)(arow + c0 + 16 + half * 8));
#pragma unroll
    for (int j = 0; j < 4; ++j) {
      const __bf16* col = McatT + (size_t)(o0 + j * 16 + m) * C_ + c0 + half * 8;
      v16bf bm = cat8(*(const v8bf*)col, *(const v8bf*)(col + 16));
      acc[j] = wmma_bf16(a, bm, acc[j]);
    }
  }

  float* zp = zpart + (size_t)cs * B_ * L_ * D_;
#pragma unroll
  for (int j = 0; j < 4; ++j)
#pragma unroll
    for (int i = 0; i < 8; ++i) {
      int t = t0 + i + half * 8;
      zp[(size_t)(b * L_ + t) * D_ + o0 + j * 16 + m] = acc[j][i];
    }
}

// z = sum of the 4 contraction partials
__global__ void k_zreduce(const float* __restrict__ zpart, float* __restrict__ z) {
  int idx = blockIdx.x * blockDim.x + threadIdx.x;
  if (idx >= B_ * L_ * D_) return;
  const size_t n = (size_t)B_ * L_ * D_;
  z[idx] = zpart[idx] + zpart[n + idx] + zpart[2 * n + idx] + zpart[3 * n + idx];
}

// --------------- distributed sequential output-AR scan ---------------------
// y_t = z_t + M_y[:,0,:] y_{t-1} + M_y[:,1,:] y_{t-2}
// 16 WGs per batch, each owns 32 rows of M_y (L2-hot 128KB slice); per-step
// release/acquire flag sync.  flags must be zeroed before launch.
__global__ void k_scan_mw(const float* __restrict__ z, const float* __restrict__ My,
                          float* __restrict__ y, int* __restrict__ flags) {
  int b   = blockIdx.x / SPLITS_;
  int g   = blockIdx.x % SPLITS_;
  int tid = threadIdx.x;          // 256
  int rowl = tid >> 3;            // 0..31: local output row
  int seg  = tid & 7;             // 0..7: 64-wide contraction segment
  int row  = g * ROWS_ + rowl;

  const float* w1 = My + (size_t)row * (KY_ * D_) + seg * 64;  // M_y[row,0,seg]
  const float* w2 = w1 + D_;                                    // M_y[row,1,seg]
  int* flg = flags + b * L_;
  float* yb = y + (size_t)b * L_ * D_;

  for (int t = 0; t < L_; ++t) {
    if (t >= 1) {
      if (tid == 0) {
        while (__hip_atomic_load(&flg[t - 1], __ATOMIC_ACQUIRE,
                                 __HIP_MEMORY_SCOPE_AGENT) < SPLITS_) {
          __builtin_amdgcn_s_sleep(1);
        }
      }
      __syncthreads();            // all threads may now read y_{t-1} (and y_{t-2})
    }
    float acc = 0.0f;
    if (t >= 1) {
      const float* y1 = yb + (size_t)(t - 1) * D_ + seg * 64;
#pragma unroll
      for (int i = 0; i < 64; i += 4) {
        float4 a = *(const float4*)(w1 + i);
        float4 p = *(const float4*)(y1 + i);
        acc += a.x * p.x + a.y * p.y + a.z * p.z + a.w * p.w;
      }
    }
    if (t >= 2) {
      const float* y2 = yb + (size_t)(t - 2) * D_ + seg * 64;
#pragma unroll
      for (int i = 0; i < 64; i += 4) {
        float4 a = *(const float4*)(w2 + i);
        float4 p = *(const float4*)(y2 + i);
        acc += a.x * p.x + a.y * p.y + a.z * p.z + a.w * p.w;
      }
    }
    // reduce the 8 segments (8-lane groups within a wave32)
    acc += __shfl_xor(acc, 4, 8);
    acc += __shfl_xor(acc, 2, 8);
    acc += __shfl_xor(acc, 1, 8);
    if (seg == 0)
      yb[(size_t)t * D_ + row] = acc + z[(size_t)(b * L_ + t) * D_ + row];
    __syncthreads();              // all rows of this WG written
    if (tid == 0) {
      __threadfence();            // make y_t rows visible device-wide
      __hip_atomic_fetch_add(&flg[t], 1, __ATOMIC_RELEASE, __HIP_MEMORY_SCOPE_AGENT);
    }
  }
}

// ------------------------------ launcher -----------------------------------

extern "C" void kernel_launch(void* const* d_in, const int* in_sizes, int n_in,
                              void* d_out, int out_size, void* d_ws, size_t ws_size,
                              hipStream_t stream) {
  const float* u  = (const float*)d_in[0];  // (B,L,D)
  const float* fl = (const float*)d_in[1];  // (L,K)
  const float* Mu = (const float*)d_in[2];  // (D,k_u,D)
  const float* My = (const float*)d_in[3];  // (D,k_y,D)
  const float* Mp = (const float*)d_in[4];  // (K,D,D)
  const float* Mm = (const float*)d_in[5];  // (K,D,D)
  float* out = (float*)d_out;               // (B,L,D) fp32

  char* ws = (char*)d_ws;
  size_t off = 0;
  __bf16* ubT   = (__bf16*)(ws + off); off += (size_t)B_ * D_ * L_ * 2;        // 2 MB
  __bf16* frev  = (__bf16*)(ws + off); off += (size_t)2 * K_ * LP_ * 2;        // 68 KB
  __bf16* McatT = (__bf16*)(ws + off); off += (size_t)D_ * C_ * 2;             // 18.4 MB
  __bf16* Ucat  = (__bf16*)(ws + off); off += (size_t)B_ * L_ * C_ * 2;        // 73.4 MB
  float*  zpart = (float*)(ws + off);  off += (size_t)CSPLIT_ * B_ * L_ * D_ * 4; // 16.8 MB
  float*  zbuf  = (float*)(ws + off);  off += (size_t)B_ * L_ * D_ * 4;        // 4.2 MB
  int*    flags = (int*)(ws + off);    off += (size_t)B_ * L_ * 4;             // 8 KB
  if (off > ws_size) return;  // workspace too small: cannot run safely

  hipMemsetAsync(flags, 0, (size_t)B_ * L_ * 4, stream);

  k_prep_ubT<<<(B_ * L_ * D_ + 255) / 256, 256, 0, stream>>>(u, ubT);
  k_prep_frev<<<(2 * K_ * LP_ + 255) / 256, 256, 0, stream>>>(fl, frev);
  k_prep_mphiT<<<(K_ * D_ * D_ + 255) / 256, 256, 0, stream>>>(Mp, Mm, McatT);
  k_prep_muT<<<(D_ * KU_ * D_ + 255) / 256, 256, 0, stream>>>(Mu, McatT);
  k_fill_shift<<<(B_ * L_ * KU_ * D_ + 255) / 256, 256, 0, stream>>>(u, Ucat);

  // Toeplitz conv: B * 2K * (L/16) * (D/64) = 32768 waves, 8 waves/block
  k_conv<<<4096, 256, 0, stream>>>(ubT, frev, Ucat);

  // fused spectral + input-AR GEMM, contraction split 4-way: 4096 waves
  k_spectral<<<512, 256, 0, stream>>>(Ucat, McatT, zpart);
  k_zreduce<<<(B_ * L_ * D_ + 255) / 256, 256, 0, stream>>>(zpart, zbuf);

  // distributed AR scan: 16 WGs per batch, flag-synced per timestep
  k_scan_mw<<<B_ * SPLITS_, 256, 0, stream>>>(zbuf, My, out, flags);
}
